// MultiHeadAttention_78752520340114
// MI455X (gfx1250) — matrix operand order repaired and checked by reading, not compiled, not run on hardware
//
#include <hip/hip_runtime.h>

typedef unsigned short u16;
typedef unsigned int   u32;
typedef __attribute__((ext_vector_type(16))) __bf16 bf16x16;
typedef __attribute__((ext_vector_type(8)))  float  f32x8;
typedef __attribute__((ext_vector_type(4)))  u32    u32x4;

union FragU { bf16x16 v; u32x4 q[2]; };

__device__ __forceinline__ u16 f2bf(float f) {
  u32 x = __float_as_uint(f);
  return (u16)((x + 0x7FFFu + ((x >> 16) & 1u)) >> 16);  // RNE
}

__device__ __forceinline__ f32x8 zero8() {
  f32x8 z;
#pragma unroll
  for (int i = 0; i < 8; ++i) z[i] = 0.0f;
  return z;
}

// A-fragment: 16x32 bf16, lane L holds row L&15.
// lane<16: K = k0+[0..7], k0+[16..23]; lane>=16: K = k0+[8..15], k0+[24..31]
__device__ __forceinline__ bf16x16 load_a(const u16* base, int row0, int ld, int k0) {
  int lane = threadIdx.x & 31;
  const u16* p = base + (size_t)(row0 + (lane & 15)) * ld + (k0 + ((lane & 16) >> 1));
  FragU f;
  f.q[0] = *(const u32x4*)(p);
  f.q[1] = *(const u32x4*)(p + 16);
  return f.v;
}

// B-fragment: 32x16 bf16, lane L holds column L&15.
// lane<16: K = k0+[0..7], k0+[16..23]; lane>=16: K = k0+[8..15], k0+[24..31]
__device__ __forceinline__ bf16x16 load_b(const u16* base, int col0, int ld, int k0) {
  int lane = threadIdx.x & 31;
  const u16* p = base + (size_t)(col0 + (lane & 15)) * ld + (k0 + ((lane & 16) >> 1));
  FragU f;
  f.q[0] = *(const u32x4*)(p);
  f.q[1] = *(const u32x4*)(p + 16);
  return f.v;
}

__device__ __forceinline__ f32x8 wmma_bf16(bf16x16 a, bf16x16 b, f32x8 c) {
  return __builtin_amdgcn_wmma_f32_16x16x32_bf16(false, a, false, b, (short)0, c,
                                                 false, false);
}

__global__ void cast_f32_bf16(const float* __restrict__ src, u16* __restrict__ dst, int n) {
  int i = blockIdx.x * blockDim.x + threadIdx.x;
  if (i < n) dst[i] = f2bf(src[i]);
}

// ---------------------------------------------------------------------------
// QKV projection: y[m,n] = sum_k x[m,k] * W[n,k]   (m = b*2048+s, n = h*64+dh)
// mode 0/1 -> store [b,h,s,dh]; mode 2 -> store V transposed [b,h,dh,s]
// Block = 4 waves, each wave owns a 64x64 output tile (16 WMMAs / K-step).
// ---------------------------------------------------------------------------
__global__ __launch_bounds__(128) void qkv_gemm(const u16* __restrict__ xb,
                                                const u16* __restrict__ wb,
                                                u16* __restrict__ dst, int mode) {
  int lane = threadIdx.x & 31;
  int wave = threadIdx.x >> 5;
  int n0 = blockIdx.x * 64;
  int m0 = blockIdx.y * 256 + wave * 64;

  f32x8 acc[4][4];
#pragma unroll
  for (int i = 0; i < 4; ++i)
#pragma unroll
    for (int j = 0; j < 4; ++j) acc[i][j] = zero8();

  for (int k = 0; k < 1024; k += 32) {
    bf16x16 af[4], bff[4];
#pragma unroll
    for (int i = 0; i < 4; ++i) af[i] = load_a(xb, m0 + 16 * i, 1024, k);
#pragma unroll
    for (int j = 0; j < 4; ++j) bff[j] = load_b(wb, n0 + 16 * j, 1024, k);
#pragma unroll
    for (int i = 0; i < 4; ++i)
#pragma unroll
      for (int j = 0; j < 4; ++j) acc[i][j] = wmma_bf16(af[i], bff[j], acc[i][j]);
  }

  int lhalf = (lane >> 4) << 3;
  int nlane = lane & 15;
#pragma unroll
  for (int i = 0; i < 4; ++i)
#pragma unroll
    for (int j = 0; j < 4; ++j)
#pragma unroll
      for (int e = 0; e < 8; ++e) {
        int m = m0 + 16 * i + e + lhalf;
        int n = n0 + 16 * j + nlane;
        int bb = m >> 11, s = m & 2047, h = n >> 6, dh = n & 63;
        size_t off = (mode == 2)
            ? (((size_t)(bb * 16 + h) * 64 + dh) * 2048 + s)    // V^T: [b,h,dh,s]
            : (((size_t)(bb * 16 + h) * 2048 + s) * 64 + dh);   // Q/K: [b,h,s,dh]
        dst[off] = f2bf(acc[i][j][e]);
      }
}

// ---------------------------------------------------------------------------
// Causal flash attention. One wave per 16-query tile per (b,h).
// Online softmax; key chunks of 32; 8 WMMAs per chunk.
// ---------------------------------------------------------------------------
__global__ __launch_bounds__(128) void flash_attn(const u16* __restrict__ qb,
                                                  const u16* __restrict__ kb,
                                                  const u16* __restrict__ vtb,
                                                  u16* __restrict__ ab) {
  __shared__ __align__(16) float sS[4][16 * 32];
  __shared__ __align__(16) u16   sP[4][16 * 32];
  __shared__ __align__(16) float sScale[4][16];

  int lane = threadIdx.x & 31;
  int wave = threadIdx.x >> 5;
  int wg = blockIdx.x * 4 + wave;
  int qt = wg & 127;          // 128 query tiles per head
  int bh = wg >> 7;           // (b*16 + h)
  int q0 = qt << 4;

  const u16* qh = qb  + (size_t)bh * 2048 * 64;
  const u16* kh = kb  + (size_t)bh * 2048 * 64;
  const u16* vh = vtb + (size_t)bh * 64 * 2048;
  float* S = sS[wave];
  u16*   P = sP[wave];
  float* SC = sScale[wave];

  int lhalf = (lane >> 4) << 3;
  int nlane = lane & 15;

  f32x8 o[4];
#pragma unroll
  for (int j = 0; j < 4; ++j) o[j] = zero8();
  float rm = -__builtin_inff();
  float rl = 0.0f;

  bf16x16 qa0 = load_a(qh, q0, 64, 0);
  bf16x16 qa1 = load_a(qh, q0, 64, 32);

  for (int kk = 0; kk < q0 + 16; kk += 32) {
    // ---- scores: S = (Q . K^T) * 1/sqrt(64), 16x32 tile ----
    f32x8 s0 = zero8(), s1 = zero8();
    bf16x16 k00 = load_b(kh, kk,      64, 0);
    bf16x16 k01 = load_b(kh, kk,      64, 32);
    bf16x16 k10 = load_b(kh, kk + 16, 64, 0);
    bf16x16 k11 = load_b(kh, kk + 16, 64, 32);
    s0 = wmma_bf16(qa0, k00, s0);
    s0 = wmma_bf16(qa1, k01, s0);
    s1 = wmma_bf16(qa0, k10, s1);
    s1 = wmma_bf16(qa1, k11, s1);

#pragma unroll
    for (int e = 0; e < 8; ++e) {
      S[(e + lhalf) * 32 + nlane]      = s0[e] * 0.125f;
      S[(e + lhalf) * 32 + 16 + nlane] = s1[e] * 0.125f;
    }
    asm volatile("s_wait_dscnt 0" ::: "memory");  // wave-local LDS visibility

    // ---- online softmax: lane r (<16) owns query row q0+r ----
    if (lane < 16) {
      int r = lane, qrow = q0 + r;
      float mx = rm;
#pragma unroll
      for (int c = 0; c < 32; ++c) {
        float v = (kk + c <= qrow) ? S[r * 32 + c] : -__builtin_inff();
        mx = fmaxf(mx, v);
      }
      float sc = __expf(rm - mx);
      float sum = 0.0f;
#pragma unroll
      for (int c = 0; c < 32; ++c) {
        float pv = (kk + c <= qrow) ? __expf(S[r * 32 + c] - mx) : 0.0f;
        sum += pv;
        P[r * 32 + c] = f2bf(pv);
      }
      rl = rl * sc + sum;
      rm = mx;
      SC[r] = sc;
    }
    asm volatile("s_wait_dscnt 0" ::: "memory");

    // ---- rescale accumulators, then O += P . V ----
#pragma unroll
    for (int j = 0; j < 4; ++j)
#pragma unroll
      for (int e = 0; e < 8; ++e) o[j][e] *= SC[e + lhalf];

    bf16x16 pa = load_a((const u16*)P, 0, 32, 0);
#pragma unroll
    for (int j = 0; j < 4; ++j) {
      bf16x16 vb = load_b(vh, j * 16, 2048, kk);   // V^T rows are contiguous in K
      o[j] = wmma_bf16(pa, vb, o[j]);
    }
  }

  // ---- normalize and scatter to [b, s, h*64+dh] (bf16 A for final GEMM) ----
  if (lane < 16) SC[lane] = 1.0f / rl;
  asm volatile("s_wait_dscnt 0" ::: "memory");

  int b = bh >> 4, h = bh & 15;
#pragma unroll
  for (int j = 0; j < 4; ++j)
#pragma unroll
    for (int e = 0; e < 8; ++e) {
      int srow = q0 + e + lhalf;
      int col = h * 64 + j * 16 + nlane;
      ab[((size_t)b * 2048 + srow) * 1024 + col] = f2bf(o[j][e] * SC[e + lhalf]);
    }
}

// ---------------------------------------------------------------------------
// Output projection with fused bias: out = attn @ W_o^T + b_o (fp32 out)
// ---------------------------------------------------------------------------
__global__ __launch_bounds__(128) void out_gemm(const u16* __restrict__ ab,
                                                const u16* __restrict__ wb,
                                                const float* __restrict__ bias,
                                                float* __restrict__ out) {
  int lane = threadIdx.x & 31;
  int wave = threadIdx.x >> 5;
  int n0 = blockIdx.x * 64;
  int m0 = blockIdx.y * 256 + wave * 64;

  f32x8 acc[4][4];
#pragma unroll
  for (int i = 0; i < 4; ++i)
#pragma unroll
    for (int j = 0; j < 4; ++j) acc[i][j] = zero8();

  for (int k = 0; k < 1024; k += 32) {
    bf16x16 af[4], bff[4];
#pragma unroll
    for (int i = 0; i < 4; ++i) af[i] = load_a(ab, m0 + 16 * i, 1024, k);
#pragma unroll
    for (int j = 0; j < 4; ++j) bff[j] = load_b(wb, n0 + 16 * j, 1024, k);
#pragma unroll
    for (int i = 0; i < 4; ++i)
#pragma unroll
      for (int j = 0; j < 4; ++j) acc[i][j] = wmma_bf16(af[i], bff[j], acc[i][j]);
  }

  int lhalf = (lane >> 4) << 3;
  int nlane = lane & 15;
#pragma unroll
  for (int i = 0; i < 4; ++i)
#pragma unroll
    for (int j = 0; j < 4; ++j)
#pragma unroll
      for (int e = 0; e < 8; ++e) {
        int m = m0 + 16 * i + e + lhalf;
        int n = n0 + 16 * j + nlane;
        out[(size_t)m * 1024 + n] = acc[i][j][e] + bias[n];
      }
}

extern "C" void kernel_launch(void* const* d_in, const int* in_sizes, int n_in,
                              void* d_out, int out_size, void* d_ws, size_t ws_size,
                              hipStream_t stream) {
  const float* x  = (const float*)d_in[0];
  const float* wq = (const float*)d_in[1];
  const float* wk = (const float*)d_in[2];
  const float* wv = (const float*)d_in[3];
  const float* wo = (const float*)d_in[4];
  const float* bo = (const float*)d_in[5];
  float* out = (float*)d_out;

  const size_t NX = (size_t)8192 * 1024;   // B*S x D
  const size_t NW = (size_t)1024 * 1024;

  u16* xb  = (u16*)d_ws;          // x, bf16
  u16* wqb = xb  + NX;
  u16* wkb = wqb + NW;
  u16* wvb = wkb + NW;
  u16* wob = wvb + NW;
  u16* qb  = wob + NW;            // Q  [b,h,s,dh]
  u16* kb  = qb  + NX;            // K  [b,h,s,dh]
  u16* vtb = kb  + NX;            // V^T [b,h,dh,s]
  u16* ab  = vtb + NX;            // attn out [b,s,d]
  // total workspace: ~92 MB (bf16)

  const int tb = 256;
  cast_f32_bf16<<<(int)((NX + tb - 1) / tb), tb, 0, stream>>>(x, xb, (int)NX);
  cast_f32_bf16<<<(int)((NW + tb - 1) / tb), tb, 0, stream>>>(wq, wqb, (int)NW);
  cast_f32_bf16<<<(int)((NW + tb - 1) / tb), tb, 0, stream>>>(wk, wkb, (int)NW);
  cast_f32_bf16<<<(int)((NW + tb - 1) / tb), tb, 0, stream>>>(wv, wvb, (int)NW);
  cast_f32_bf16<<<(int)((NW + tb - 1) / tb), tb, 0, stream>>>(wo, wob, (int)NW);

  dim3 g(16, 32);
  qkv_gemm<<<g, 128, 0, stream>>>(xb, wqb, qb, 0);
  qkv_gemm<<<g, 128, 0, stream>>>(xb, wkb, kb, 1);
  qkv_gemm<<<g, 128, 0, stream>>>(xb, wvb, vtb, 2);

  flash_attn<<<2048, 128, 0, stream>>>(qb, kb, vtb, ab);

  out_gemm<<<g, 128, 0, stream>>>(ab, wob, bo, out);
}
